// DGCNNModel_69423851373236
// MI455X (gfx1250) — matrix operand order repaired and checked by reading, not compiled, not run on hardware
//
#include <hip/hip_runtime.h>

// ---------------------------------------------------------------------------
// DGCNN forward for MI455X (gfx1250, wave32, WMMA bf16).
// EdgeConv is refactored as  out[i,o] = max_k lrelu( (x_j.w1)[o] + (x_i.(w2-w1))[o] )
// so every heavy op is a dense bf16 WMMA GEMM; kNN distance tiles live in LDS.
// ---------------------------------------------------------------------------

#define BATCH 8
#define NPTS  2048
#define KNN   20
#define PSEG  150
#define DDIM  512
#define COUT  50

typedef __attribute__((ext_vector_type(16))) __bf16 bf16x16;
typedef __attribute__((ext_vector_type(8)))  __bf16 bf16x8;
typedef __attribute__((ext_vector_type(8)))  float  f32x8;

// ---- WMMA fragment loaders (row-major bf16 sources) -----------------------
// A: 16x32 tile (M x K). Lane l<16: row=l, elems 0..15 -> K k0..k0+15.
// Lane l>=16: same rows, elems 0..15 -> K k0+16..k0+31.
__device__ __forceinline__ bf16x16 load_a_frag(const __bf16* __restrict__ X,
                                               int lda, int row0, int k0, int lane) {
    int r  = row0 + (lane & 15);
    int kb = k0 + ((lane & 16) ? 16 : 0);
    const __bf16* p = X + (long)r * lda + kb;
    bf16x8 lo = *(const bf16x8*)p;
    bf16x8 hi = *(const bf16x8*)(p + 8);
    bf16x16 a;
#pragma unroll
    for (int i = 0; i < 8; ++i) { a[i] = lo[i]; a[8 + i] = hi[i]; }
    return a;
}
// B: 32x16 tile (K x N) taken as rows of weight W (N x K):  B[k,n] = W[col0+n, k].
// Lanes 0..15 hold K k0..k0+15, lanes 16..31 hold K k0+16..k0+31 (contiguous 32B).
__device__ __forceinline__ bf16x16 load_b_frag(const __bf16* __restrict__ W,
                                               int ldw, int col0, int k0, int lane) {
    int c  = col0 + (lane & 15);
    int kb = k0 + ((lane & 16) ? 16 : 0);
    return *(const bf16x16*)(W + (long)c * ldw + kb);
}

__device__ __forceinline__ float lrelu(float x) { return x >= 0.f ? x : 0.2f * x; }

// ---------------------------------------------------------------------------
// Generic bf16 WMMA GEMM:  out = act( A(MxK) * W(NxK)^T + bias )
// Block = 256 threads (8 waves); block tile 32 rows x 256 cols;
// wave tile = 16 rows x 64 cols (4 accumulators, A frag reused 4x via cache).
// ---------------------------------------------------------------------------
__global__ __launch_bounds__(256)
void gemm_bf16_kernel(const __bf16* __restrict__ A, int lda,
                      const __bf16* __restrict__ W, int ldw,
                      float* outF, __bf16* outB, int ldo,
                      const float* __restrict__ bias,
                      int M, int N, int Npad, int K, int act) {
    int lane = threadIdx.x & 31;
    int w    = threadIdx.x >> 5;
    int rowTile = blockIdx.x * 32 + (w & 1) * 16;
    int colBase = blockIdx.y * 256 + (w >> 1) * 64;
    if (rowTile + 16 > M) return;  // wave-uniform

    for (int cg = 0; cg < 4; ++cg) {
        int col0 = colBase + cg * 16;
        if (col0 + 16 > Npad) continue;  // wave-uniform
        f32x8 acc = {0.f, 0.f, 0.f, 0.f, 0.f, 0.f, 0.f, 0.f};
        for (int k0 = 0; k0 < K; k0 += 32) {
            bf16x16 a = load_a_frag(A, lda, rowTile, k0, lane);
            bf16x16 b = load_b_frag(W, ldw, col0, k0, lane);
            acc = __builtin_amdgcn_wmma_f32_16x16x32_bf16(
                false, a, false, b, (short)0, acc, false, false);
        }
        int col = col0 + (lane & 15);
        bool cok = col < N;
        float bs = (bias != nullptr && cok) ? bias[col] : 0.f;
#pragma unroll
        for (int v = 0; v < 8; ++v) {
            int row = rowTile + v + ((lane & 16) ? 8 : 0);
            float x = acc[v] + bs;
            if (act) x = lrelu(x);
            if (cok) {
                long o = (long)row * ldo + col;
                if (outF) outF[o] = x;
                if (outB) outB[o] = (__bf16)x;
            }
        }
    }
}

// ---------------------------------------------------------------------------
// kNN: one block per (batch, 16-query tile). Phase 1: WMMA inner products vs
// all 2048 points; full 16x2048 fp32 distance tile kept in LDS (320KB/WGP on
// CDNA5). Phase 2: 20 rounds of lane-strided argmin + wave32 shuffle reduce.
// ---------------------------------------------------------------------------
#define LDD 2056  // 2048 + 8 floats padding: row stride offsets banks by 8

template <int KP>
__global__ __launch_bounds__(256)
void knn_kernel(const __bf16* __restrict__ X, int ldx,
                const float* __restrict__ xx, int* __restrict__ idx_out) {
    extern __shared__ float sdist[];
    constexpr int NKS = KP / 32;
    int lane = threadIdx.x & 31;
    int w    = threadIdx.x >> 5;          // 0..7
    int b    = blockIdx.x >> 7;           // NPTS/16 = 128 tiles per batch
    int qt   = blockIdx.x & 127;
    int qbase = qt * 16;
    const __bf16* Xb = X + (long)b * NPTS * ldx;
    const float* xxb = xx + b * NPTS;

    // preload query-block A fragments (registers)
    bf16x16 afr[NKS];
#pragma unroll
    for (int ks = 0; ks < NKS; ++ks)
        afr[ks] = load_a_frag(Xb, ldx, qbase, ks * 32, lane);

    // query norms for this lane's output rows
    float qn[8];
#pragma unroll
    for (int v = 0; v < 8; ++v) qn[v] = xxb[qbase + v + ((lane & 16) ? 8 : 0)];

    // each wave covers 256 columns = 16 column tiles
    for (int t = 0; t < 16; ++t) {
        int c0 = w * 256 + t * 16;
        f32x8 acc = {0.f, 0.f, 0.f, 0.f, 0.f, 0.f, 0.f, 0.f};
#pragma unroll
        for (int ks = 0; ks < NKS; ++ks) {
            bf16x16 bf = load_b_frag(Xb, ldx, c0, ks * 32, lane);
            acc = __builtin_amdgcn_wmma_f32_16x16x32_bf16(
                false, afr[ks], false, bf, (short)0, acc, false, false);
        }
        int col = c0 + (lane & 15);
        float xc = xxb[col];
#pragma unroll
        for (int v = 0; v < 8; ++v) {
            int lr = v + ((lane & 16) ? 8 : 0);
            sdist[lr * LDD + col] = qn[v] + xc - 2.f * acc[v];
        }
    }
    __syncthreads();

    // top-20 selection: wave w owns local rows 2w and 2w+1
    for (int rr = 0; rr < 2; ++rr) {
        int lr = w * 2 + rr;
        float* drow = sdist + lr * LDD;
        int grow = qbase + lr;
        for (int s = 0; s < KNN; ++s) {
            float best = 3.4e38f; int bidx = 0x7fffffff;
            for (int j = lane; j < NPTS; j += 32) {
                float v = drow[j];
                if (v < best) { best = v; bidx = j; }
            }
#pragma unroll
            for (int off = 16; off > 0; off >>= 1) {
                float ov = __shfl_xor(best, off, 32);
                int   oi = __shfl_xor(bidx, off, 32);
                if (ov < best || (ov == best && oi < bidx)) { best = ov; bidx = oi; }
            }
            if (lane == 0) {
                drow[bidx] = 3.4e38f;
                idx_out[((long)b * NPTS + grow) * KNN + s] = bidx;
            }
        }
    }
}

// ---------------------------------------------------------------------------
// Small data-movement / pointwise kernels
// ---------------------------------------------------------------------------
__global__ void cast_src_kernel(const float* __restrict__ src, __bf16* __restrict__ x0b) {
    int i = blockIdx.x * blockDim.x + threadIdx.x;
    if (i >= BATCH * NPTS) return;
    const float* s = src + (long)i * 3;
    __bf16* d = x0b + (long)i * 32;
    d[0] = (__bf16)s[0]; d[1] = (__bf16)s[1]; d[2] = (__bf16)s[2];
    for (int c = 3; c < 32; ++c) d[c] = (__bf16)0.f;
}

// build [w1 ; w2-w1] (2O x Kp) bf16 from conv weight (O x 2C) fp32
__global__ void prep_w_kernel(const float* __restrict__ w, int O, int C, int Kp,
                              __bf16* __restrict__ out) {
    int t = blockIdx.x * blockDim.x + threadIdx.x;
    int total = 2 * O * Kp;
    if (t >= total) return;
    int k = t % Kp, r = t / Kp;
    float v = 0.f;
    if (k < C) {
        if (r < O) v = w[(long)r * 2 * C + k];
        else { int rr = r - O; v = w[(long)rr * 2 * C + C + k] - w[(long)rr * 2 * C + k]; }
    }
    out[t] = (__bf16)v;
}

__global__ void cast_w_kernel(const float* __restrict__ w, int Nrows, int Npad, int K,
                              __bf16* __restrict__ out) {
    int t = blockIdx.x * blockDim.x + threadIdx.x;
    if (t >= Npad * K) return;
    int k = t % K, r = t / K;
    out[t] = (__bf16)(r < Nrows ? w[(long)r * K + k] : 0.f);
}

__global__ void norm_kernel(const __bf16* __restrict__ X, int ldx, int C,
                            float* __restrict__ xx) {
    int i = blockIdx.x * blockDim.x + threadIdx.x;
    if (i >= BATCH * NPTS) return;
    const __bf16* p = X + (long)i * ldx;
    float s = 0.f;
    for (int c = 0; c < C; ++c) { float f = (float)p[c]; s += f * f; }
    xx[i] = s;
}

// out[i,o] = max_k lrelu( Y[idx_k, o] + Y[i, O+o] ), written bf16 into xcat slice
__global__ void edge_reduce_kernel(const float* __restrict__ Y, int O,
                                   const int* __restrict__ idx,
                                   __bf16* __restrict__ xcat, int ldx, int coff) {
    int t = blockIdx.x * blockDim.x + threadIdx.x;
    int total = BATCH * NPTS * O;
    if (t >= total) return;
    int o = t % O;
    int bi = t / O;
    int b = bi / NPTS;
    int twoO = 2 * O;
    float t2 = Y[(long)bi * twoO + O + o];
    const int* ip = idx + (long)bi * KNN;
    float m = -3.4e38f;
    for (int k = 0; k < KNN; ++k) {
        long j = (long)b * NPTS + ip[k];
        m = fmaxf(m, lrelu(Y[j * twoO + o] + t2));
    }
    xcat[(long)bi * ldx + coff + o] = (__bf16)m;
}

// monotone float<->uint encoding so segment-max maps to native atomicMax(u32)
__device__ __forceinline__ unsigned encf(float f) {
    unsigned u = __float_as_uint(f);
    return (u & 0x80000000u) ? ~u : (u | 0x80000000u);
}
__device__ __forceinline__ float decf(unsigned u) {
    unsigned v = (u & 0x80000000u) ? (u & 0x7fffffffu) : ~u;
    return __uint_as_float(v);
}

__global__ void seg_init_kernel(unsigned* __restrict__ segu, float* __restrict__ cnt,
                                int* __restrict__ maxseg) {
    int t = blockIdx.x * blockDim.x + threadIdx.x;
    if (t < BATCH * PSEG * DDIM) segu[t] = encf(-3.4e38f);
    if (t < BATCH * PSEG) cnt[t] = 0.f;
    if (t < BATCH) maxseg[t] = 0;
}

__global__ void seg_scatter_kernel(const float* __restrict__ feat,
                                   const int* __restrict__ segments,
                                   unsigned* __restrict__ segu,
                                   float* __restrict__ cnt, int* __restrict__ maxseg) {
    int t = blockIdx.x * blockDim.x + threadIdx.x;
    if (t >= BATCH * NPTS * DDIM) return;
    int d = t % DDIM;
    int bi = t / DDIM;
    int b = bi / NPTS;
    int s = segments[bi];
    atomicMax(&segu[((long)b * PSEG + s) * DDIM + d], encf(feat[t]));
    if (d == 0) { atomicAdd(&cnt[b * PSEG + s], 1.f); atomicMax(&maxseg[b], s); }
}

__global__ void gmax_kernel(const float* __restrict__ feat, float* __restrict__ gf) {
    int t = blockIdx.x * blockDim.x + threadIdx.x;
    if (t >= BATCH * DDIM) return;
    int b = t / DDIM, d = t % DDIM;
    const float* p = feat + (long)b * NPTS * DDIM + d;
    float m = -3.4e38f;
    for (int i = 0; i < NPTS; ++i) m = fmaxf(m, p[(long)i * DDIM]);
    gf[t] = m;
}

__global__ void fsim_kernel(const unsigned* __restrict__ segu, const float* __restrict__ cnt,
                            const int* __restrict__ maxseg, const float* __restrict__ gf,
                            __bf16* __restrict__ fsimb) {
    int t = blockIdx.x * blockDim.x + threadIdx.x;
    if (t >= BATCH * PSEG * 2 * DDIM) return;
    int d2 = t % (2 * DDIM);
    int bp = t / (2 * DDIM);
    int b = bp / PSEG, p = bp % PSEG;
    float v = 0.f;
    if (p <= maxseg[b]) {
        if (d2 < DDIM) v = (cnt[bp] > 0.f) ? decf(segu[(long)bp * DDIM + d2]) : 0.f;
        else v = gf[b * DDIM + (d2 - DDIM)];
    }
    fsimb[t] = (__bf16)v;
}

__global__ void out_kernel(const float* __restrict__ o3, float* __restrict__ out) {
    int t = blockIdx.x * blockDim.x + threadIdx.x;
    if (t >= BATCH * COUT * PSEG) return;
    int p = t % PSEG;
    int r = t / PSEG;
    int o = r % COUT, b = r / COUT;
    out[t] = o3[((long)b * PSEG + p) * 64 + o];
}

// ---------------------------------------------------------------------------
// Host orchestration
// ---------------------------------------------------------------------------
static inline unsigned cdiv(long a, long b) { return (unsigned)((a + b - 1) / b); }

static void gemm(const __bf16* A, int lda, const __bf16* W, int ldw,
                 float* outF, __bf16* outB, int ldo, const float* bias,
                 int M, int N, int Npad, int K, int act, hipStream_t s) {
    dim3 g(cdiv(M, 32), cdiv(Npad, 256));
    gemm_bf16_kernel<<<g, 256, 0, s>>>(A, lda, W, ldw, outF, outB, ldo, bias,
                                       M, N, Npad, K, act);
}

extern "C" void kernel_launch(void* const* d_in, const int* in_sizes, int n_in,
                              void* d_out, int out_size, void* d_ws, size_t ws_size,
                              hipStream_t stream) {
    const float* src      = (const float*)d_in[0];
    const int*   segments = (const int*)d_in[2];
    const float* c1w = (const float*)d_in[3];
    const float* c2w = (const float*)d_in[4];
    const float* c3w = (const float*)d_in[5];
    const float* c4w = (const float*)d_in[6];
    const float* c5w = (const float*)d_in[7];
    const float* m1w = (const float*)d_in[8];
    const float* m2w = (const float*)d_in[9];
    const float* m2b = (const float*)d_in[10];
    const float* m3w = (const float*)d_in[11];
    const float* m3b = (const float*)d_in[12];
    float* out = (float*)d_out;

    const int M = BATCH * NPTS;   // 16384
    const int MP = BATCH * PSEG;  // 1200

    // --- carve workspace (256B aligned) ---
    char* base = (char*)d_ws;
    size_t off = 0;
    auto carve = [&](size_t bytes) {
        void* p = base + off;
        off = (off + bytes + 255) & ~(size_t)255;
        return p;
    };
    __bf16*   x0b   = (__bf16*)carve((size_t)M * 32 * 2);
    float*    xx    = (float*)carve((size_t)M * 4);
    int*      idx   = (int*)carve((size_t)M * KNN * 4);
    float*    Y     = (float*)carve((size_t)M * 512 * 4);
    __bf16*   xcat  = (__bf16*)carve((size_t)M * 512 * 2);
    float*    feat  = (float*)carve((size_t)M * 512 * 4);
    __bf16*   wc1   = (__bf16*)carve((size_t)128 * 32 * 2);
    __bf16*   wc2   = (__bf16*)carve((size_t)128 * 64 * 2);
    __bf16*   wc3   = (__bf16*)carve((size_t)256 * 64 * 2);
    __bf16*   wc4   = (__bf16*)carve((size_t)512 * 128 * 2);
    __bf16*   w5b   = (__bf16*)carve((size_t)512 * 512 * 2);
    __bf16*   m1b   = (__bf16*)carve((size_t)512 * 1024 * 2);
    __bf16*   m2bw  = (__bf16*)carve((size_t)256 * 512 * 2);
    __bf16*   m3bw  = (__bf16*)carve((size_t)64 * 256 * 2);
    float*    gf    = (float*)carve((size_t)BATCH * DDIM * 4);
    unsigned* segu  = (unsigned*)carve((size_t)BATCH * PSEG * DDIM * 4);
    float*    cnt   = (float*)carve((size_t)BATCH * PSEG * 4);
    int*      mseg  = (int*)carve((size_t)BATCH * 4);
    __bf16*   fsimb = (__bf16*)carve((size_t)MP * 1024 * 2);
    __bf16*   h1b   = (__bf16*)carve((size_t)MP * 512 * 2);
    __bf16*   h2b   = (__bf16*)carve((size_t)MP * 256 * 2);
    float*    o3    = (float*)carve((size_t)MP * 64 * 4);
    (void)ws_size; (void)n_in; (void)in_sizes; (void)out_size;

    const int T = 256;
    const int KNN_LDS = 16 * LDD * 4;  // ~129KB distance tile (CDNA5: 320KB/WGP)

    // --- weight preparation (bf16, EdgeConv weights split as [w1; w2-w1]) ---
    cast_src_kernel<<<cdiv(M, T), T, 0, stream>>>(src, x0b);
    prep_w_kernel<<<cdiv(128 * 32, T), T, 0, stream>>>(c1w, 64, 3, 32, wc1);
    prep_w_kernel<<<cdiv(128 * 64, T), T, 0, stream>>>(c2w, 64, 64, 64, wc2);
    prep_w_kernel<<<cdiv(256 * 64, T), T, 0, stream>>>(c3w, 128, 64, 64, wc3);
    prep_w_kernel<<<cdiv(512 * 128, T), T, 0, stream>>>(c4w, 256, 128, 128, wc4);
    cast_w_kernel<<<cdiv(512 * 512, T), T, 0, stream>>>(c5w, 512, 512, 512, w5b);
    cast_w_kernel<<<cdiv(512 * 1024, T), T, 0, stream>>>(m1w, 512, 512, 1024, m1b);
    cast_w_kernel<<<cdiv(256 * 512, T), T, 0, stream>>>(m2w, 256, 256, 512, m2bw);
    cast_w_kernel<<<cdiv(64 * 256, T), T, 0, stream>>>(m3w, 50, 64, 256, m3bw);

    // --- EdgeConv layer 1 (Cin=3 padded to 32, O=64 -> xcat[:, 0:64]) ---
    norm_kernel<<<cdiv(M, T), T, 0, stream>>>(x0b, 32, 32, xx);
    knn_kernel<32><<<M / 16, 256, KNN_LDS, stream>>>(x0b, 32, xx, idx);
    gemm(x0b, 32, wc1, 32, Y, nullptr, 128, nullptr, M, 128, 128, 32, 0, stream);
    edge_reduce_kernel<<<cdiv((long)M * 64, T), T, 0, stream>>>(Y, 64, idx, xcat, 512, 0);

    // --- layer 2 (Cin=64 from xcat[:,0:64], O=64 -> xcat[:, 64:128]) ---
    norm_kernel<<<cdiv(M, T), T, 0, stream>>>(xcat, 512, 64, xx);
    knn_kernel<64><<<M / 16, 256, KNN_LDS, stream>>>(xcat, 512, xx, idx);
    gemm(xcat, 512, wc2, 64, Y, nullptr, 128, nullptr, M, 128, 128, 64, 0, stream);
    edge_reduce_kernel<<<cdiv((long)M * 64, T), T, 0, stream>>>(Y, 64, idx, xcat, 512, 64);

    // --- layer 3 (Cin=64 from xcat[:,64:128], O=128 -> xcat[:, 128:256]) ---
    norm_kernel<<<cdiv(M, T), T, 0, stream>>>(xcat + 64, 512, 64, xx);
    knn_kernel<64><<<M / 16, 256, KNN_LDS, stream>>>(xcat + 64, 512, xx, idx);
    gemm(xcat + 64, 512, wc3, 64, Y, nullptr, 256, nullptr, M, 256, 256, 64, 0, stream);
    edge_reduce_kernel<<<cdiv((long)M * 128, T), T, 0, stream>>>(Y, 128, idx, xcat, 512, 128);

    // --- layer 4 (Cin=128 from xcat[:,128:256], O=256 -> xcat[:, 256:512]) ---
    norm_kernel<<<cdiv(M, T), T, 0, stream>>>(xcat + 128, 512, 128, xx);
    knn_kernel<128><<<M / 16, 256, KNN_LDS, stream>>>(xcat + 128, 512, xx, idx);
    gemm(xcat + 128, 512, wc4, 128, Y, nullptr, 512, nullptr, M, 512, 512, 128, 0, stream);
    edge_reduce_kernel<<<cdiv((long)M * 256, T), T, 0, stream>>>(Y, 256, idx, xcat, 512, 256);

    // --- conv5: feat = lrelu(xcat @ conv5_w^T), fp32 (B,N,512) ---
    gemm(xcat, 512, w5b, 512, feat, nullptr, 512, nullptr, M, 512, 512, 512, 1, stream);

    // --- global max + ragged segment max + fsim (B,P,1024) bf16 ---
    seg_init_kernel<<<cdiv((long)BATCH * PSEG * DDIM, T), T, 0, stream>>>(segu, cnt, mseg);
    seg_scatter_kernel<<<cdiv((long)M * DDIM, T), T, 0, stream>>>(feat, segments, segu, cnt, mseg);
    gmax_kernel<<<cdiv(BATCH * DDIM, T), T, 0, stream>>>(feat, gf);
    fsim_kernel<<<cdiv((long)MP * 1024, T), T, 0, stream>>>(segu, cnt, mseg, gf, fsimb);

    // --- MLP head: 1024 -> 512 -> 256 -> 50 (M = B*P = 1200) ---
    gemm(fsimb, 1024, m1b, 1024, nullptr, h1b, 512, nullptr, MP, 512, 512, 1024, 1, stream);
    gemm(h1b, 512, m2bw, 512, nullptr, h2b, 256, m2b, MP, 256, 256, 512, 1, stream);
    gemm(h2b, 256, m3bw, 256, o3, nullptr, 64, m3b, MP, 50, 64, 256, 0, stream);

    // --- (B,P,50) -> (B,50,P) ---
    out_kernel<<<cdiv(BATCH * COUT * PSEG, T), T, 0, stream>>>(o3, out);
}
